// MultiModalModel_82970178224176
// MI455X (gfx1250) — matrix operand order repaired and checked by reading, not compiled, not run on hardware
//
#include <hip/hip_runtime.h>

// ---------------------------------------------------------------------------
// Fused cross-batch token-similarity + logsumexp pooling for MI455X (gfx1250).
//   clip_sims[i,j] = mean_n( TAU * logsumexp_m( <A[i,n,:],V[j,m,:]> / TAU ) )
// B=32, Na=512, Nv=256, D=384, fp32 in/out.  ~103 GFLOP fp32-equivalent GEMM,
// inputs fit in L2 -> matrix-core bound.  Strategy: hi/lo bf16 split of fp32
// operands -> 3x v_wmma_f32_16x16x32_bf16 per tile (near-fp32 accuracy at bf16
// matrix rate).  Operands pre-split once into workspace in LDS-ready tiled
// layout; hot loop stages tiles with async global->LDS copies (ASYNCcnt).
// ---------------------------------------------------------------------------

typedef __attribute__((ext_vector_type(8)))  __bf16 v8bf;
typedef __attribute__((ext_vector_type(16))) __bf16 v16bf;
typedef __attribute__((ext_vector_type(8)))  float  v8f;
typedef __attribute__((ext_vector_type(4)))  int    v4i;
typedef __attribute__((address_space(1))) v4i* v4i_gp;   // global 16B chunk
typedef __attribute__((address_space(3))) v4i* v4i_lp;   // LDS    16B chunk

#define BB   32
#define NA   512
#define NV   256
#define DD   384
#define TAU  2.0f

#define M_TILE 64
#define RB     (NA / M_TILE)   // 8 row blocks per (i,j) pair
#define KSTEP  32
#define NKS    (DD / KSTEP)    // 12 k-steps
#define LDA    40              // padded row stride (bf16 elems) = 80 bytes
#define SLD    257             // padded row stride (floats) for S tile
#define SMEM_BYTES 65792       // max( staging 51200 B , S tile 64*257*4 B )

// Packed workspace geometry (bf16 bit patterns as ushort, pad baked in):
//   A block  (per i,rb,ks): [hi 64x40 | lo 64x40]   = 5120  ush = 10240 B
//   V block  (per j,ks)   : [hi 256x40| lo 256x40]  = 20480 ush = 40960 B
#define ABLK_USH 5120
#define VBLK_USH 20480
#define APK_BYTES ((size_t)BB * RB * NKS * ABLK_USH * 2)   // 31,457,280
#define VPK_BYTES ((size_t)BB * NKS * VBLK_USH * 2)        // 15,728,640
#define PART_BYTES ((size_t)BB * BB * RB * 4)              // 32,768
#define WS_NEEDED (PART_BYTES + APK_BYTES + VPK_BYTES)

#if defined(__HIP_DEVICE_COMPILE__) && __has_builtin(__builtin_amdgcn_global_load_async_to_lds_b128)
#define ASYNC_LDS 1
#else
#define ASYNC_LDS 0
#endif

// ---- fp32 -> bf16 (round-to-nearest-even) helpers -------------------------
__device__ __forceinline__ unsigned short bf16_rne(float x) {
    unsigned int u = __float_as_uint(x);
    unsigned int r = u + 0x7FFFu + ((u >> 16) & 1u);
    return (unsigned short)(r >> 16);
}

__device__ __forceinline__ void split_store4(unsigned short* __restrict__ hp,
                                             unsigned short* __restrict__ lp,
                                             float4 v) {
    unsigned short h[4], l[4];
    const float* f = &v.x;
#pragma unroll
    for (int e = 0; e < 4; ++e) {
        unsigned short hb = bf16_rne(f[e]);
        float hf = __uint_as_float(((unsigned int)hb) << 16);
        h[e] = hb;
        l[e] = bf16_rne(f[e] - hf);   // residual -> near-fp32 overall precision
    }
    *(ushort4*)hp = make_ushort4(h[0], h[1], h[2], h[3]);
    *(ushort4*)lp = make_ushort4(l[0], l[1], l[2], l[3]);
}

// ---- 16B global -> LDS copy (async path when available) --------------------
__device__ __forceinline__ void cp16(unsigned short* lds, const unsigned short* g) {
#if ASYNC_LDS
    __builtin_amdgcn_global_load_async_to_lds_b128((v4i_gp)g, (v4i_lp)lds, 0, 0);
#else
    *(float4*)(void*)lds = *(const float4*)(const void*)g;
#endif
}

__device__ __forceinline__ void async_wait0() {
#if ASYNC_LDS
#if __has_builtin(__builtin_amdgcn_s_wait_asynccnt)
    __builtin_amdgcn_s_wait_asynccnt(0);
#else
    asm volatile("s_wait_asynccnt 0x0" ::: "memory");
#endif
#endif
}

// ---- WMMA fragment loads from LDS (ISA 7.12.2 16-bit layouts) --------------
// A (16x32, MxK): lane<16 -> row=lane, K {0..7,16..23}; lane>=16 -> K {8..15,24..31}
__device__ __forceinline__ v16bf ld_afrag(const unsigned short* p) {
    v8bf a = *(const v8bf*)p;
    v8bf b = *(const v8bf*)(p + 16);
    return __builtin_shufflevector(a, b, 0,1,2,3,4,5,6,7,8,9,10,11,12,13,14,15);
}
// B (32x16, KxN), stored N-major: lane<16 -> col=lane, K {0..7,16..23}; lane>=16 -> K {8..15,24..31}
__device__ __forceinline__ v16bf ld_bfrag(const unsigned short* p) {
    v8bf a = *(const v8bf*)p;
    v8bf b = *(const v8bf*)(p + 16);
    return __builtin_shufflevector(a, b, 0,1,2,3,4,5,6,7,8,9,10,11,12,13,14,15);
}

// ---- one-time hi/lo split into LDS-ready tiled layout ----------------------
__global__ __launch_bounds__(256)
void pack_audio_kernel(const float* __restrict__ a, unsigned short* __restrict__ Apk) {
    int g = blockIdx.x * 256 + threadIdx.x;        // 1,572,864 float4's
    int base = g * 4;
    int i   = base / (NA * DD);
    int rem = base - i * (NA * DD);
    int n   = rem / DD;
    int d   = rem - n * DD;
    float4 v = *(const float4*)(a + base);
    int rb = n >> 6, r = n & 63, ks = d >> 5, c = d & 31;
    unsigned short* hp = Apk + (size_t)((i * RB + rb) * NKS + ks) * ABLK_USH + r * LDA + c;
    split_store4(hp, hp + 2560, v);                // lo plane = +64*40
}

__global__ __launch_bounds__(256)
void pack_visual_kernel(const float* __restrict__ vsl, unsigned short* __restrict__ Vpk) {
    int g = blockIdx.x * 256 + threadIdx.x;        // 786,432 float4's
    int base = g * 4;
    int j   = base / (NV * DD);
    int rem = base - j * (NV * DD);
    int m   = rem / DD;
    int d   = rem - m * DD;
    float4 v = *(const float4*)(vsl + base);
    int ks = d >> 5, c = d & 31;
    unsigned short* hp = Vpk + (size_t)(j * NKS + ks) * VBLK_USH + m * LDA + c;
    split_store4(hp, hp + 10240, v);               // lo plane = +256*40
}

// ---- main fused kernel -----------------------------------------------------
template <bool PACKED>
__global__ __launch_bounds__(256)
void fused_sim_lse_kernel(const float* __restrict__ audio,
                          const float* __restrict__ visual,
                          const unsigned short* __restrict__ Apk,
                          const unsigned short* __restrict__ Vpk,
                          float* __restrict__ partials) {
    __shared__ __align__(16) unsigned char smem[SMEM_BYTES];
    unsigned short* As_hi = (unsigned short*)(smem);             //  64 x 40 bf16
    unsigned short* As_lo = (unsigned short*)(smem + 5120);
    unsigned short* Vs_hi = (unsigned short*)(smem + 10240);     // 256 x 40 bf16
    unsigned short* Vs_lo = (unsigned short*)(smem + 30720);
    float* S = (float*)smem;                                     // 64 x 257 f32 (aliased)

    const int tid  = threadIdx.x;
    const int lane = tid & 31;
    const int wave = tid >> 5;
    const int m_wave = wave & 1;   // 2 row-groups of 32
    const int n_wave = wave >> 1;  // 4 col-groups of 64

    const int bx   = blockIdx.x;                 // B*B*RB = 8192 workgroups
    const int rb   = bx & (RB - 1);
    const int pair = bx >> 3;
    const int i    = pair >> 5;
    const int j    = pair & (BB - 1);
    const int rowbase = rb * M_TILE;

    v8f acc[2][4];
#pragma unroll
    for (int mm = 0; mm < 2; ++mm)
#pragma unroll
        for (int nn = 0; nn < 4; ++nn)
#pragma unroll
            for (int e = 0; e < 8; ++e) acc[mm][nn][e] = 0.0f;

    for (int ks = 0; ks < NKS; ++ks) {
        if constexpr (PACKED) {
            // ---- linear copy of pre-split tiles (51200 B) into LDS ---------
            const unsigned short* ag = Apk + (size_t)((i * RB + rb) * NKS + ks) * ABLK_USH;
            const unsigned short* vg = Vpk + (size_t)(j * NKS + ks) * VBLK_USH;
            unsigned short* lb = (unsigned short*)smem;
#pragma unroll
            for (int u = 0; u < 3; ++u) {          // A: 640 x 16B chunks
                int idx = tid + u * 256;
                if (idx < 640) cp16(lb + idx * 8, ag + idx * 8);
            }
#pragma unroll
            for (int u = 0; u < 10; ++u) {         // V: 2560 x 16B chunks
                int idx = tid + u * 256;
                cp16(lb + 5120 + idx * 8, vg + idx * 8);
            }
            async_wait0();
        } else {
            // ---- fallback: load fp32, hi/lo split on the fly ---------------
            const int k0 = ks * KSTEP;
#pragma unroll
            for (int u = 0; u < 2; ++u) {
                int idx = tid + u * 256;
                int r = idx >> 3, c4 = (idx & 7) * 4;
                const float4 v = *(const float4*)(audio +
                    (size_t)(i * NA + rowbase + r) * DD + k0 + c4);
                split_store4(As_hi + r * LDA + c4, As_lo + r * LDA + c4, v);
            }
#pragma unroll
            for (int u = 0; u < 8; ++u) {
                int idx = tid + u * 256;
                int r = idx >> 3, c4 = (idx & 7) * 4;
                const float4 v = *(const float4*)(visual +
                    (size_t)(j * NV + r) * DD + k0 + c4);
                split_store4(Vs_hi + r * LDA + c4, Vs_lo + r * LDA + c4, v);
            }
        }
        __syncthreads();

        // ---- per-wave fragments + WMMA (hi*hi + hi*lo + lo*hi) -------------
        v16bf ahi[2], alo[2];
#pragma unroll
        for (int mm = 0; mm < 2; ++mm) {
            const int ao = (m_wave * 32 + mm * 16 + (lane & 15)) * LDA + ((lane >> 4) * 8);
            ahi[mm] = ld_afrag(As_hi + ao);
            alo[mm] = ld_afrag(As_lo + ao);
        }
#pragma unroll
        for (int nn = 0; nn < 4; ++nn) {
            const int bo = (n_wave * 64 + nn * 16 + (lane & 15)) * LDA + ((lane >> 4) * 8);
            v16bf bhi = ld_bfrag(Vs_hi + bo);
            v16bf blo = ld_bfrag(Vs_lo + bo);
#pragma unroll
            for (int mm = 0; mm < 2; ++mm) {
                acc[mm][nn] = __builtin_amdgcn_wmma_f32_16x16x32_bf16(
                    false, ahi[mm], false, bhi, (short)0, acc[mm][nn], false, false);
                acc[mm][nn] = __builtin_amdgcn_wmma_f32_16x16x32_bf16(
                    false, ahi[mm], false, blo, (short)0, acc[mm][nn], false, false);
                acc[mm][nn] = __builtin_amdgcn_wmma_f32_16x16x32_bf16(
                    false, alo[mm], false, bhi, (short)0, acc[mm][nn], false, false);
            }
        }
        __syncthreads();
    }

    // ---- spill S tile (64x256) to LDS; C/D layout: lanes 0-15 M=e, 16-31 M=e+8
#pragma unroll
    for (int mm = 0; mm < 2; ++mm)
#pragma unroll
        for (int nn = 0; nn < 4; ++nn) {
            const int row0 = m_wave * 32 + mm * 16 + ((lane >> 4) * 8);
            const int col  = n_wave * 64 + nn * 16 + (lane & 15);
#pragma unroll
            for (int e = 0; e < 8; ++e)
                S[(row0 + e) * SLD + col] = acc[mm][nn][e];
        }
    __syncthreads();

    // ---- row-wise tau*logsumexp (max-stabilized), then block-partial mean ----
    if (tid < M_TILE) {
        const float* rowp = S + tid * SLD;
        float m = rowp[0];
#pragma unroll 4
        for (int c = 1; c < NV; ++c) m = fmaxf(m, rowp[c]);
        float s = 0.0f;
#pragma unroll 4
        for (int c = 0; c < NV; ++c) s += __expf((rowp[c] - m) * (1.0f / TAU));
        S[tid * SLD + NV] = m + TAU * __logf(s);   // stash in pad slot
    }
    __syncthreads();
    if (tid == 0) {
        float t = 0.0f;
        for (int r = 0; r < M_TILE; ++r) t += S[r * SLD + NV];
        partials[bx] = t * (1.0f / (float)NA);
    }
}

__global__ __launch_bounds__(256)
void reduce_partials_kernel(const float* __restrict__ partials,
                            float* __restrict__ out) {
    int g = blockIdx.x * blockDim.x + threadIdx.x;
    if (g < BB * BB) {
        float s = 0.0f;
#pragma unroll
        for (int r = 0; r < RB; ++r) s += partials[g * RB + r];
        out[g] = s;
    }
}

extern "C" void kernel_launch(void* const* d_in, const int* in_sizes, int n_in,
                              void* d_out, int out_size, void* d_ws, size_t ws_size,
                              hipStream_t stream) {
    (void)in_sizes; (void)n_in; (void)out_size;
    const float* audio  = (const float*)d_in[0];   // (32, 512, 384) fp32
    const float* visual = (const float*)d_in[1];   // (32, 256, 384) fp32
    float* out      = (float*)d_out;               // (32, 32) fp32
    float* partials = (float*)d_ws;                // first 32 KB of workspace

    if (ws_size >= WS_NEEDED) {
        unsigned short* Apk = (unsigned short*)((char*)d_ws + PART_BYTES);
        unsigned short* Vpk = (unsigned short*)((char*)d_ws + PART_BYTES + APK_BYTES);
        pack_audio_kernel<<<dim3(6144), dim3(256), 0, stream>>>(audio, Apk);
        pack_visual_kernel<<<dim3(3072), dim3(256), 0, stream>>>(visual, Vpk);
        fused_sim_lse_kernel<true><<<dim3(BB * BB * RB), dim3(256), 0, stream>>>(
            audio, visual, Apk, Vpk, partials);
    } else {
        fused_sim_lse_kernel<false><<<dim3(BB * BB * RB), dim3(256), 0, stream>>>(
            audio, visual, nullptr, nullptr, partials);
    }
    reduce_partials_kernel<<<dim3((BB * BB + 255) / 256), dim3(256), 0, stream>>>(
        partials, out);
}
